// feature_extraction_70557722739462
// MI455X (gfx1250) — 8-bit matrix operand order repaired and checked by reading, compiled, not run on hardware
//
#include <hip/hip_runtime.h>
#include <hip/hip_bf16.h>
#include <stdint.h>

typedef __attribute__((ext_vector_type(8))) int v8i;
typedef __attribute__((ext_vector_type(4))) int v4i;

#define BN_EPS 1e-5f
#define NOCT 2

__global__ void prep_sb_kernel(const float* __restrict__ w, const float* __restrict__ gamma,
                               const float* __restrict__ beta, const float* __restrict__ mean,
                               const float* __restrict__ var, float* __restrict__ scale,
                               float* __restrict__ bias, int OC, int cnt, int use_alpha)
{
    int oc = blockIdx.x * blockDim.x + threadIdx.x;
    if (oc >= OC) return;
    float a = 1.0f;
    if (use_alpha) {
        float s = 0.0f;
        const float* wp = w + (size_t)oc * cnt;
        for (int i = 0; i < cnt; ++i) s += fabsf(wp[i]);
        a = s / (float)cnt;
    }
    float inv = gamma[oc] * rsqrtf(var[oc] + BN_EPS);
    scale[oc] = a * inv;
    bias[oc]  = beta[oc] - mean[oc] * inv;
}

__global__ void pack_w_kernel(const float* __restrict__ w, int8_t* __restrict__ wpk,
                              int OC, int Cin, int K, int Cpad, int nK)
{
    int bi = blockIdx.x * blockDim.x + threadIdx.x;
    int total = (OC >> 4) * nK * 1024;
    if (bi >= total) return;
    int octile = bi / (nK * 1024);
    int r  = bi % (nK * 1024);
    int kt = r >> 10;
    int q  = r & 1023;
    int lane = q >> 5;
    int p  = q & 31;
    int M  = lane & 15, hi = lane >> 4;
    int v  = p >> 3,    rr = p & 7;
    int k  = kt * 64 + (v >> 1) * 32 + hi * 16 + (v & 1) * 8 + rr;
    int tap = k / Cpad, ci = k % Cpad;
    int oc  = octile * 16 + M;
    int8_t byte = 0;
    if (ci < Cin) {
        int ky = tap / K, kx = tap % K;
        float wv = w[(((size_t)oc * Cin + ci) * K + ky) * K + kx];
        byte = (wv >= 0.0f) ? (int8_t)1 : (int8_t)-1;
    }
    wpk[bi] = byte;
}

__global__ void encode_kernel(const float* __restrict__ x, int8_t* __restrict__ out, int npix)
{
    int p = blockIdx.x * blockDim.x + threadIdx.x;
    if (p >= npix) return;
    int q = (int)rintf(x[p] * 255.0f);
    q = q < 0 ? 0 : (q > 255 ? 255 : q);
    char b[8];
#pragma unroll
    for (int k = 0; k < 8; ++k) b[k] = ((q >> k) & 1) ? 1 : -1;
    int64_t lv; __builtin_memcpy(&lv, b, 8);
    *(int64_t*)(out + (size_t)p * 64) = lv;
}

__global__ void sign_kernel(const float* __restrict__ in, int8_t* __restrict__ out,
                            int HW, int C, int Cpad, int total)
{
    int i = blockIdx.x * blockDim.x + threadIdx.x;
    if (i >= total) return;
    int pcol = i % HW;
    int c = (i / HW) % C;
    int b = i / (HW * C);
    out[((size_t)(b * HW + pcol)) * Cpad + c] = (in[i] >= 0.0f) ? 1 : -1;
}

__global__ void up2_kernel(const float* __restrict__ in, float* __restrict__ out,
                           int BC, int Hin, int Win)
{
    int Hout = Hin * 2, Wout = Win * 2;
    int total = BC * Hout * Wout;
    int i = blockIdx.x * blockDim.x + threadIdx.x;
    if (i >= total) return;
    int x = i % Wout, y = (i / Wout) % Hout, bc = i / (Wout * Hout);
    float fy = (y + 0.5f) * 0.5f - 0.5f;
    float fx = (x + 0.5f) * 0.5f - 0.5f;
    int y0 = (int)floorf(fy), x0 = (int)floorf(fx);
    float ty = fy - (float)y0, tx = fx - (float)x0;
    int y1 = y0 + 1, x1 = x0 + 1;
    y0 = max(0, min(Hin - 1, y0)); y1 = max(0, min(Hin - 1, y1));
    x0 = max(0, min(Win - 1, x0)); x1 = max(0, min(Win - 1, x1));
    const float* p = in + (size_t)bc * Hin * Win;
    float v00 = p[y0 * Win + x0], v01 = p[y0 * Win + x1];
    float v10 = p[y1 * Win + x0], v11 = p[y1 * Win + x1];
    out[i] = (1.f - ty) * ((1.f - tx) * v00 + tx * v01) + ty * ((1.f - tx) * v10 + tx * v11);
}

__global__ void copy_ch_kernel(const int8_t* __restrict__ src, int8_t* __restrict__ dst,
                               int npix, int C, int Cs, int Cd, int coff)
{
    int total = npix * C;
    int i = blockIdx.x * blockDim.x + threadIdx.x;
    if (i >= total) return;
    int p = i / C, c = i % C;
    dst[(size_t)p * Cd + coff + c] = src[(size_t)p * Cs + c];
}

__global__ void down_conv_kernel(const float* __restrict__ x, const float* __restrict__ w,
                                 const float* __restrict__ scale, const float* __restrict__ bias,
                                 float* __restrict__ out, int Bn, int Cin, int Hin, int Win,
                                 int OC, int Hout, int Wout, int stride)
{
    int total = Bn * OC * Hout * Wout;
    int i = blockIdx.x * blockDim.x + threadIdx.x;
    if (i >= total) return;
    int xo = i % Wout, yo = (i / Wout) % Hout;
    int oc = (i / (Wout * Hout)) % OC, b = i / (Wout * Hout * OC);
    const float* xp = x + (((size_t)b * Cin) * Hin + yo * stride) * Win + xo * stride;
    const float* wp = w + (size_t)oc * Cin;
    float s = 0.0f;
    for (int ci = 0; ci < Cin; ++ci) s += xp[(size_t)ci * Hin * Win] * wp[ci];
    out[i] = s * scale[oc] + bias[oc];
}

__global__ void __launch_bounds__(256) bconv_wmma_kernel(
    const int8_t* __restrict__ act, const int8_t* __restrict__ wpk,
    const float* __restrict__ scale, const float* __restrict__ bias,
    const float* __restrict__ res, float* __restrict__ fout, int8_t* __restrict__ sout,
    int Bn, int Hin, int Win, int Cpad,
    int Hout, int Wout, int stride, int pad, int Kk,
    int fC, int fcoff, int resC, int rescoff, int sC, int relu)
{
    extern __shared__ int4 smem4[];
    int8_t* smem = (int8_t*)smem4;

    const int nKc = Cpad >> 6;
    const int octile0 = blockIdx.y * NOCT;
    const int wslab = (Kk * Kk * nKc) << 10;

    {
        const int4* wsrc = (const int4*)(wpk + (size_t)octile0 * wslab);
        int n16 = (NOCT * wslab) >> 4;
        for (int i = threadIdx.x; i < n16; i += blockDim.x) smem4[i] = wsrc[i];
    }
    __syncthreads();

    const int lane = threadIdx.x & 31;
    const int wave = threadIdx.x >> 5;
    const int hi = lane >> 4;
    const int ln = lane & 15;
    const int HW = Hout * Wout;
    const int npix = Bn * HW;
    const int tile = blockIdx.x * 8 + wave;
    const bool wvalid = (tile * 16 < npix);
    int p = wvalid ? (tile * 16 + ln) : ln;
    int b = p / HW, r0 = p % HW;
    int oy = r0 / Wout, ox = r0 % Wout;
    int iy0 = oy * stride - pad, ix0 = ox * stride - pad;

    const int rowstride = Win * Cpad;
    const int tapbytes = nKc << 10;
    const int8_t* arow = smem + lane * 32;

    v8i acc0 = {0, 0, 0, 0, 0, 0, 0, 0};
    v8i acc1 = {0, 0, 0, 0, 0, 0, 0, 0};

    for (int dy = 0; dy < Kk; ++dy) {
        int iy = iy0 + dy;
        bool rowok = (iy >= 0) & (iy < Hin);
        const int8_t* rowbase =
            act + ((size_t)(b * Hin + iy)) * rowstride + (size_t)ix0 * Cpad + (hi << 4);
        if (rowok)
            __builtin_prefetch((const void*)(rowbase + rowstride), 0, 1);
        for (int dx = 0; dx < Kk; ++dx) {
            int ix = ix0 + dx;
            bool inb = rowok & (ix >= 0) & (ix < Win);
            const int8_t* bbase = rowbase + dx * Cpad;
            for (int kc = 0; kc < nKc; ++kc) {
                v4i blo = {0, 0, 0, 0}, bh = {0, 0, 0, 0};
                if (inb) {
                    const v4i* bp = (const v4i*)(bbase + (kc << 6));
                    blo = bp[0];
                    bh  = bp[2];
                }
                v8i bfrag = {blo[0], blo[1], blo[2], blo[3], bh[0], bh[1], bh[2], bh[3]};
                const int8_t* ab = arow + (kc << 10);
                {
                    const v4i* ap = (const v4i*)ab;
                    v4i a0 = ap[0], a1 = ap[1];
                    v8i afrag = {a0[0], a0[1], a0[2], a0[3], a1[0], a1[1], a1[2], a1[3]};
                    acc0 = __builtin_amdgcn_wmma_i32_16x16x64_iu8(true, afrag, true, bfrag,
                                                                  acc0, false, false);
                }
                {
                    const v4i* ap = (const v4i*)(ab + wslab);
                    v4i a0 = ap[0], a1 = ap[1];
                    v8i afrag = {a0[0], a0[1], a0[2], a0[3], a1[0], a1[1], a1[2], a1[3]};
                    acc1 = __builtin_amdgcn_wmma_i32_16x16x64_iu8(true, afrag, true, bfrag,
                                                                  acc1, false, false);
                }
            }
            arow += tapbytes;
        }
    }

    if (!wvalid) return;

    v8i accs[NOCT] = {acc0, acc1};
#pragma unroll
    for (int t = 0; t < NOCT; ++t) {
        const int ocbase = (octile0 + t) * 16 + hi * 8;
        float vals[8];
#pragma unroll
        for (int rI = 0; rI < 8; ++rI) {
            int oc = ocbase + rI;
            float v = (float)accs[t][rI] * scale[oc] + bias[oc];
            if (res)  v += res[(((size_t)b * resC + rescoff + oc) * Hout + oy) * Wout + ox];
            if (relu) v = v > 0.0f ? v : 0.0f;
            vals[rI] = v;
            if (fout) fout[(((size_t)b * fC + fcoff + oc) * Hout + oy) * Wout + ox] = v;
        }
        if (sout) {
            char bytes[8];
#pragma unroll
            for (int rI = 0; rI < 8; ++rI) bytes[rI] = (vals[rI] >= 0.0f) ? 1 : -1;
            int64_t lv; __builtin_memcpy(&lv, bytes, 8);
            *(int64_t*)(sout + (size_t)p * sC + ocbase) = lv;
        }
    }
}

static inline int ceildiv(int a, int b) { return (a + b - 1) / b; }

enum { G_C1A = 0, G_C2A, G_C3A, G_FC0, G_FC1, G_FC2,
       G_I1C1, G_I1C2, G_I2C1, G_I2C2, G_I3C1, G_I3C2,
       G_L1C1, G_L1C2, G_L2C1, G_L2C2, G_L2D,
       G_L3C1, G_L3C2, G_L3D, G_L4C1, G_L4C2, G_RED, NG };

static const int OCt[NG]   = { 96,128,160, 64, 64, 32,  64, 64, 96, 96,128,128,
                               32, 32, 64, 64, 64, 128,128,128, 128,128, 64 };
static const int CINt[NG]  = { 64, 96,128,  8, 64, 64,  96,128,128,192,160,256,
                               32, 32, 32, 64, 32,  64,128, 64, 128,128,128 };
static const int Kt[NG]    = {  3,  3,  3,  7,  5,  3,   3,  3,  3,  3,  3,  3,
                                3,  3,  3,  3,  1,   3,  3,  1,   3,  3,  3 };
static const int CPADt[NG] = { 64,128,128, 64, 64, 64, 128,128,128,192,192,256,
                               64, 64, 64, 64,  0,  64,128,  0, 128,128,128 };

extern "C" void kernel_launch(void* const* d_in, const int* in_sizes, int n_in,
                              void* d_out, int out_size, void* d_ws, size_t ws_size,
                              hipStream_t stream)
{
    (void)in_sizes; (void)n_in; (void)out_size; (void)ws_size;
    const float* x = (const float*)d_in[0];
    float* out = (float*)d_out;
    float* gwc = out + 7864320;

    int8_t* wsb = (int8_t*)d_ws;
    size_t off = 0;
    auto alloc = [&](size_t n) -> int8_t* {
        int8_t* p = wsb + off; off = (off + n + 255) & ~(size_t)255; return p;
    };

    int8_t* b_enc  = alloc((size_t)2*512*960*64);
    int8_t* b0     = alloc((size_t)2*512*960*64);
    int8_t* b1     = alloc((size_t)2*256*480*64);
    int8_t* b2     = alloc((size_t)2*256*480*64);
    int8_t* b3     = alloc((size_t)2*256*480*64);
    int8_t* b4     = alloc((size_t)2*256*480*64);
    int8_t* b5     = alloc((size_t)2*128*240*64);
    int8_t* b6     = alloc((size_t)2*128*240*64);
    int8_t* b7     = alloc((size_t)2*64*120*128);
    int8_t* b8     = alloc((size_t)2*64*120*128);
    int8_t* b9     = alloc((size_t)2*64*120*128);
    int8_t* b10    = alloc((size_t)2*64*120*128);
    int8_t* b11    = alloc((size_t)2*64*120*64);
    int8_t* b12    = alloc((size_t)2*32*60*128);
    int8_t* b13    = alloc((size_t)2*16*30*128);
    int8_t* b_u3   = alloc((size_t)2*16*30*192);
    int8_t* b_cat3 = alloc((size_t)2*16*30*256);
    int8_t* b_u2   = alloc((size_t)2*32*60*128);
    int8_t* b_cat2 = alloc((size_t)2*32*60*192);
    int8_t* b_u1   = alloc((size_t)2*64*120*128);
    int8_t* b_cat1 = alloc((size_t)2*64*120*128);
    size_t int8_end = off;

    auto allocf = [&](size_t n) -> float* {
        float* p = (float*)(wsb + off); off = (off + 4*n + 255) & ~(size_t)255; return p;
    };
    float* f2     = allocf(7864320);
    float* f_res  = allocf(3932160);
    float* f6     = allocf(3932160);
    float* f_res2 = allocf(1966080);
    float* f_c3   = allocf(38400);
    float* f_u3   = allocf(153600);
    float* f_i3   = allocf(122880);
    float* f_u2   = allocf(491520);
    float* f_i2   = allocf(368640);
    float* f_u1   = allocf(1474560);

    float* scaleA[NG]; float* biasA[NG]; int8_t* wpkA[NG];
    for (int g = 0; g < NG; ++g) { scaleA[g] = allocf(OCt[g]); biasA[g] = allocf(OCt[g]); wpkA[g] = nullptr; }
    for (int g = 0; g < NG; ++g) {
        if (!CPADt[g]) continue;
        int nK = Kt[g]*Kt[g]*(CPADt[g] >> 6);
        wpkA[g] = alloc((size_t)(OCt[g] >> 4) * nK * 1024);
    }

    hipMemsetAsync(wsb, 0, int8_end, stream);

    for (int g = 0; g < NG; ++g) {
        const float* beta  = (const float*)d_in[1 + 5*g + 0];
        const float* gamma = (const float*)d_in[1 + 5*g + 1];
        const float* mean  = (const float*)d_in[1 + 5*g + 2];
        const float* var   = (const float*)d_in[1 + 5*g + 3];
        const float* w     = (const float*)d_in[1 + 5*g + 4];
        int OC = OCt[g], Cin = CINt[g], K = Kt[g];
        prep_sb_kernel<<<ceildiv(OC, 64), 64, 0, stream>>>(
            w, gamma, beta, mean, var, scaleA[g], biasA[g], OC, Cin*K*K, CPADt[g] ? 1 : 0);
        if (CPADt[g]) {
            int nK = K*K*(CPADt[g] >> 6);
            int total = (OC >> 4) * nK * 1024;
            pack_w_kernel<<<ceildiv(total, 256), 256, 0, stream>>>(
                w, wpkA[g], OC, Cin, K, CPADt[g], nK);
        }
    }

    encode_kernel<<<ceildiv(983040, 256), 256, 0, stream>>>(x, b_enc, 983040);

    auto bconv = [&](int g, const int8_t* in, int Hin, int Win, int stride,
                     const float* resp, int resC, int rescoff,
                     float* foutp, int fC, int fcoff,
                     int8_t* soutp, int sC, int relu) {
        int OC = OCt[g], K = Kt[g], Cpad = CPADt[g];
        int pad = K >> 1;
        int Hout = (Hin + 2*pad - K) / stride + 1;
        int Wout = (Win + 2*pad - K) / stride + 1;
        int nK = K*K*(Cpad >> 6);
        int npix = 2 * Hout * Wout;
        dim3 grid((unsigned)ceildiv(npix / 16, 8), (unsigned)(OC / (16 * NOCT)));
        bconv_wmma_kernel<<<grid, 256, (size_t)NOCT * nK * 1024, stream>>>(
            in, wpkA[g], scaleA[g], biasA[g], resp, foutp, soutp,
            2, Hin, Win, Cpad, Hout, Wout, stride, pad, K,
            fC, fcoff, resC, rescoff, sC, relu);
    };

    bconv(G_FC0, b_enc, 512, 960, 1, nullptr,0,0, nullptr,0,0, b0, 64, 1);
    bconv(G_FC1, b0,    512, 960, 2, nullptr,0,0, nullptr,0,0, b1, 64, 1);
    bconv(G_FC2, b1,    256, 480, 1, nullptr,0,0, f2, 32, 0,   b2, 64, 1);
    bconv(G_L1C1, b2,   256, 480, 1, nullptr,0,0, nullptr,0,0, b3, 64, 1);
    bconv(G_L1C2, b3,   256, 480, 1, f2, 32, 0,   out, 32, 0,  b4, 64, 1);
    {
        const float* w = (const float*)d_in[1 + 5*G_L2D + 4];
        down_conv_kernel<<<ceildiv(2*64*128*240, 256), 256, 0, stream>>>(
            out, w, scaleA[G_L2D], biasA[G_L2D], f_res, 2, 32, 256, 480, 64, 128, 240, 2);
    }
    bconv(G_L2C1, b4, 256, 480, 2, nullptr,0,0, nullptr,0,0,  b5, 64, 1);
    bconv(G_L2C2, b5, 128, 240, 1, f_res, 64, 0, f6, 64, 0,   b6, 64, 1);
    {
        const float* w = (const float*)d_in[1 + 5*G_L3D + 4];
        down_conv_kernel<<<ceildiv(2*128*64*120, 256), 256, 0, stream>>>(
            f6, w, scaleA[G_L3D], biasA[G_L3D], f_res2, 2, 64, 128, 240, 128, 64, 120, 2);
    }
    bconv(G_L3C1, b6, 128, 240, 2, nullptr,0,0,   nullptr,0,0,    b7, 128, 1);
    bconv(G_L3C2, b7,  64, 120, 1, f_res2,128,0,  gwc, 384, 0,    b8, 128, 1);
    bconv(G_L4C1, b8,  64, 120, 1, nullptr,0,0,   nullptr,0,0,    b9, 128, 1);
    bconv(G_L4C2, b9,  64, 120, 1, gwc, 384, 0,   gwc, 384, 128,  b10,128, 1);
    bconv(G_RED,  b10, 64, 120, 1, nullptr,0,0,   gwc, 384, 256,  b11, 64, 1);
    bconv(G_C1A,  b11, 64, 120, 2, nullptr,0,0,   nullptr,0,0,    b12,128, 1);
    bconv(G_C2A,  b12, 32,  60, 2, nullptr,0,0,   nullptr,0,0,    b13,128, 1);
    bconv(G_C3A,  b13, 16,  30, 2, nullptr,0,0,   f_c3, 160, 0,   nullptr,0, 1);

    up2_kernel <<<ceildiv(2*160*16*30, 256), 256, 0, stream>>>(f_c3, f_u3, 2*160, 8, 15);
    sign_kernel<<<ceildiv(2*160*16*30, 256), 256, 0, stream>>>(f_u3, b_u3, 16*30, 160, 192, 2*160*16*30);
    bconv(G_I3C1, b_u3, 16, 30, 1, nullptr,0,0, nullptr,0,0, b_cat3, 256, 1);
    copy_ch_kernel<<<ceildiv(2*16*30*128, 256), 256, 0, stream>>>(b13, b_cat3, 2*16*30, 128, 128, 256, 128);
    bconv(G_I3C2, b_cat3, 16, 30, 1, nullptr,0,0, f_i3, 128, 0, nullptr, 0, 1);

    up2_kernel <<<ceildiv(2*128*32*60, 256), 256, 0, stream>>>(f_i3, f_u2, 2*128, 16, 30);
    sign_kernel<<<ceildiv(2*128*32*60, 256), 256, 0, stream>>>(f_u2, b_u2, 32*60, 128, 128, 2*128*32*60);
    bconv(G_I2C1, b_u2, 32, 60, 1, nullptr,0,0, nullptr,0,0, b_cat2, 192, 1);
    copy_ch_kernel<<<ceildiv(2*32*60*96, 256), 256, 0, stream>>>(b12, b_cat2, 2*32*60, 96, 128, 192, 96);
    bconv(G_I2C2, b_cat2, 32, 60, 1, nullptr,0,0, f_i2, 96, 0, nullptr, 0, 1);

    up2_kernel <<<ceildiv(2*96*64*120, 256), 256, 0, stream>>>(f_i2, f_u1, 2*96, 32, 60);
    sign_kernel<<<ceildiv(2*96*64*120, 256), 256, 0, stream>>>(f_u1, b_u1, 64*120, 96, 128, 2*96*64*120);
    bconv(G_I1C1, b_u1, 64, 120, 1, nullptr,0,0, nullptr,0,0, b_cat1, 128, 1);
    copy_ch_kernel<<<ceildiv(2*64*120*64, 256), 256, 0, stream>>>(b11, b_cat1, 2*64*120, 64, 64, 128, 64);
    bconv(G_I1C2, b_cat1, 64, 120, 1, nullptr,0,0, gwc, 384, 320, nullptr, 0, 1);
}
